// FlashACE_79422535237752
// MI455X (gfx1250) — matrix operand order repaired and checked by reading, not compiled, not run on hardware
//
#include <hip/hip_runtime.h>

#define N_NODES 50000
#define N_EDGES 1600000
#define HID 128
#define REST 96
#define OUTW (HID + REST)   // 224
#define CAT 257             // 2*HID + 1
#define NCH1 9              // ceil(257/32)
#define NCH2 4              // 128/32

typedef __attribute__((ext_vector_type(16))) __bf16 v16bf;
typedef __attribute__((ext_vector_type(8)))  float  v8f;
typedef __attribute__((ext_vector_type(4)))  unsigned int u32x4;

union Frag {
    v16bf v;
    u32x4 q[2];
    unsigned short u[16];
};

__device__ __forceinline__ unsigned short f2bf(float f) {
    unsigned int x = __float_as_uint(f);
    unsigned int r = 0x7FFFu + ((x >> 16) & 1u);
    return (unsigned short)((x + r) >> 16);
}

// ---------------------------------------------------------------------------
// out = h (full 224-wide copy; rest columns are final, scalars are working set)
__global__ void copy_h_kernel(const float* __restrict__ h, float* __restrict__ out, int n) {
    int i = blockIdx.x * blockDim.x + threadIdx.x;
    if (i < n) out[i] = h[i];
}

// ---------------------------------------------------------------------------
// Pre-shuffle a row-major fp32 weight [K x 128] into fragment-ready bf16
// B-operand layout for v_wmma_f32_16x16x32_bf16:
//   dst[((c*8 + t)*32 + lane)*16 + e]  where lane half selects K runs.
__global__ void frag_build_kernel(const float* __restrict__ W, int Kvalid, int nChunks,
                                  unsigned short* __restrict__ dst) {
    int gid = blockIdx.x * blockDim.x + threadIdx.x;
    int total = nChunks * 8 * 32;
    if (gid >= total) return;
    int lane = gid & 31;
    int t    = (gid >> 5) & 7;
    int c    = gid >> 8;
    int half = lane >> 4;
    int n    = t * 16 + (lane & 15);
    int kb   = c * 32;
    unsigned short* o = dst + (size_t)gid * 16;
#pragma unroll
    for (int e = 0; e < 16; ++e) {
        int k = (e < 8) ? (kb + 8 * half + e) : (kb + 16 + 8 * half + (e - 8));
        float v = (k < Kvalid) ? W[(size_t)k * HID + n] : 0.0f;
        o[e] = f2bf(v);
    }
}

// ---------------------------------------------------------------------------
// Per layer: bf16 node-feature table + residual-initialized fp32 accumulator
__global__ void init_layer_kernel(const float* __restrict__ out,
                                  unsigned short* __restrict__ sc16,
                                  float* __restrict__ agg) {
    int i = blockIdx.x * blockDim.x + threadIdx.x;
    if (i >= N_NODES * HID) return;
    int n = i >> 7;
    int c = i & (HID - 1);
    float v = out[(size_t)n * OUTW + c];
    sc16[i] = f2bf(v);
    agg[i]  = v;
}

__global__ void writeback_kernel(const float* __restrict__ agg, float* __restrict__ out) {
    int i = blockIdx.x * blockDim.x + threadIdx.x;
    if (i >= N_NODES * HID) return;
    int n = i >> 7;
    int c = i & (HID - 1);
    out[(size_t)n * OUTW + c] = agg[i];
}

// ---------------------------------------------------------------------------
// Edge message-passing: one wave = 16 edges. Two back-to-back bf16 WMMA GEMMs
// (16x288 @ 288x128 -> silu -> 16x128 @ 128x128) then f32 atomic segment-sum.
__global__ __launch_bounds__(256) void edge_mp_kernel(
    const unsigned short* __restrict__ sc16,
    const int* __restrict__ sender, const int* __restrict__ recv,
    const float* __restrict__ elen,
    const unsigned short* __restrict__ w1fr, const unsigned short* __restrict__ w2fr,
    const float* __restrict__ b1, const float* __restrict__ b2,
    float* __restrict__ agg) {
    __shared__ __align__(16) unsigned short hid[8][16 * HID];

    int lane = threadIdx.x & 31;
    int wave = threadIdx.x >> 5;
    int tile = blockIdx.x * 8 + wave;
    long base = (long)tile * 16;
    int half = lane >> 4;
    int mcol = lane & 15;

    int nodeS = sender[base + mcol];
    int nodeR = recv[base + mcol];

    // --- A fragments: 4 chunks sender feats, 4 chunks receiver feats, 1 len chunk
    Frag a[NCH1];
#pragma unroll
    for (int c = 0; c < 4; ++c) {
        const unsigned short* rp = sc16 + (size_t)nodeS * HID + c * 32 + 8 * half;
        a[c].q[0] = *(const u32x4*)(rp);
        a[c].q[1] = *(const u32x4*)(rp + 16);
    }
#pragma unroll
    for (int c = 0; c < 4; ++c) {
        const unsigned short* rp = sc16 + (size_t)nodeR * HID + c * 32 + 8 * half;
        a[4 + c].q[0] = *(const u32x4*)(rp);
        a[4 + c].q[1] = *(const u32x4*)(rp + 16);
    }
#pragma unroll
    for (int e = 0; e < 16; ++e) a[8].u[e] = 0;
    if (half == 0) a[8].u[0] = f2bf(elen[base + mcol]);   // K=256 column

    // --- GEMM1 (+bias) -> SiLU -> stage bf16 hidden tile in LDS
    unsigned short* hw = hid[wave];
#pragma unroll
    for (int t = 0; t < 8; ++t) {
        float bv = b1[t * 16 + mcol];
        v8f acc;
#pragma unroll
        for (int r = 0; r < 8; ++r) acc[r] = bv;
#pragma unroll
        for (int c = 0; c < NCH1; ++c) {
            Frag b;
            const unsigned short* wp = w1fr + ((size_t)(c * 8 + t) * 32 + lane) * 16;
            b.q[0] = *(const u32x4*)(wp);
            b.q[1] = *(const u32x4*)(wp + 8);
            acc = __builtin_amdgcn_wmma_f32_16x16x32_bf16(
                false, a[c].v, false, b.v, (short)0, acc, false, false);
        }
#pragma unroll
        for (int r = 0; r < 8; ++r) {
            float x = acc[r];
            float s = x * (1.0f / (1.0f + __expf(-x)));
            hw[(r + 8 * half) * HID + t * 16 + mcol] = f2bf(s);
        }
    }
    __syncthreads();   // uniform: exact grid, all waves arrive

    // --- A fragments for GEMM2 from the LDS hidden tile
    Frag a2[NCH2];
#pragma unroll
    for (int c = 0; c < NCH2; ++c) {
        const unsigned short* rp = hw + mcol * HID + c * 32 + 8 * half;
        a2[c].q[0] = *(const u32x4*)(rp);
        a2[c].q[1] = *(const u32x4*)(rp + 16);
    }

    // receiver node for each C-tile row (M = r + 8*half)
    int rowNode[8];
#pragma unroll
    for (int r = 0; r < 8; ++r) rowNode[r] = __shfl(nodeR, r + 8 * half, 32);

    // --- GEMM2 (+bias) -> atomic segment-sum into agg
#pragma unroll
    for (int t = 0; t < 8; ++t) {
        float bv = b2[t * 16 + mcol];
        v8f acc;
#pragma unroll
        for (int r = 0; r < 8; ++r) acc[r] = 0.0f;
#pragma unroll
        for (int c = 0; c < NCH2; ++c) {
            Frag b;
            const unsigned short* wp = w2fr + ((size_t)(c * 8 + t) * 32 + lane) * 16;
            b.q[0] = *(const u32x4*)(wp);
            b.q[1] = *(const u32x4*)(wp + 8);
            acc = __builtin_amdgcn_wmma_f32_16x16x32_bf16(
                false, a2[c].v, false, b.v, (short)0, acc, false, false);
        }
#pragma unroll
        for (int r = 0; r < 8; ++r) {
            atomicAdd(agg + (size_t)rowNode[r] * HID + t * 16 + mcol, acc[r] + bv);
        }
    }
}

// ---------------------------------------------------------------------------
// Node update: scalars += silu(scalars@nu_w1+b1)@nu_w2+b2.
// Single-wave workgroups (3125 tiles exactly) -> no cross-wave barriers needed.
__global__ __launch_bounds__(32) void node_update_kernel(
    const unsigned short* __restrict__ sc16,
    const unsigned short* __restrict__ nu1fr, const unsigned short* __restrict__ nu2fr,
    const float* __restrict__ b1, const float* __restrict__ b2,
    float* __restrict__ out) {
    __shared__ __align__(16) unsigned short hw[16 * HID];

    int lane = threadIdx.x & 31;
    int tile = blockIdx.x;
    int half = lane >> 4;
    int mcol = lane & 15;
    int node = tile * 16 + mcol;

    Frag a[NCH2];
#pragma unroll
    for (int c = 0; c < NCH2; ++c) {
        const unsigned short* rp = sc16 + (size_t)node * HID + c * 32 + 8 * half;
        a[c].q[0] = *(const u32x4*)(rp);
        a[c].q[1] = *(const u32x4*)(rp + 16);
    }

#pragma unroll
    for (int t = 0; t < 8; ++t) {
        float bv = b1[t * 16 + mcol];
        v8f acc;
#pragma unroll
        for (int r = 0; r < 8; ++r) acc[r] = bv;
#pragma unroll
        for (int c = 0; c < NCH2; ++c) {
            Frag b;
            const unsigned short* wp = nu1fr + ((size_t)(c * 8 + t) * 32 + lane) * 16;
            b.q[0] = *(const u32x4*)(wp);
            b.q[1] = *(const u32x4*)(wp + 8);
            acc = __builtin_amdgcn_wmma_f32_16x16x32_bf16(
                false, a[c].v, false, b.v, (short)0, acc, false, false);
        }
#pragma unroll
        for (int r = 0; r < 8; ++r) {
            float x = acc[r];
            float s = x * (1.0f / (1.0f + __expf(-x)));
            hw[(r + 8 * half) * HID + t * 16 + mcol] = f2bf(s);
        }
    }

    Frag a2[NCH2];
#pragma unroll
    for (int c = 0; c < NCH2; ++c) {
        const unsigned short* rp = hw + mcol * HID + c * 32 + 8 * half;
        a2[c].q[0] = *(const u32x4*)(rp);
        a2[c].q[1] = *(const u32x4*)(rp + 16);
    }

#pragma unroll
    for (int t = 0; t < 8; ++t) {
        float bv = b2[t * 16 + mcol];
        v8f acc;
#pragma unroll
        for (int r = 0; r < 8; ++r) acc[r] = 0.0f;
#pragma unroll
        for (int c = 0; c < NCH2; ++c) {
            Frag b;
            const unsigned short* wp = nu2fr + ((size_t)(c * 8 + t) * 32 + lane) * 16;
            b.q[0] = *(const u32x4*)(wp);
            b.q[1] = *(const u32x4*)(wp + 8);
            acc = __builtin_amdgcn_wmma_f32_16x16x32_bf16(
                false, a2[c].v, false, b.v, (short)0, acc, false, false);
        }
#pragma unroll
        for (int r = 0; r < 8; ++r) {
            size_t idx = (size_t)(tile * 16 + r + 8 * half) * OUTW + t * 16 + mcol;
            out[idx] = out[idx] + acc[r] + bv;   // residual
        }
    }
}

// ---------------------------------------------------------------------------
extern "C" void kernel_launch(void* const* d_in, const int* in_sizes, int n_in,
                              void* d_out, int out_size, void* d_ws, size_t ws_size,
                              hipStream_t stream) {
    const float* h       = (const float*)d_in[0];
    const int*   eidx    = (const int*)d_in[1];
    const float* elen    = (const float*)d_in[2];
    const float* mp_w1   = (const float*)d_in[3];
    const float* mp_b1   = (const float*)d_in[4];
    const float* mp_w2   = (const float*)d_in[5];
    const float* mp_b2   = (const float*)d_in[6];
    const float* nu_w1   = (const float*)d_in[7];
    const float* nu_b1   = (const float*)d_in[8];
    const float* nu_w2   = (const float*)d_in[9];
    const float* nu_b2   = (const float*)d_in[10];
    float* out = (float*)d_out;

    const int* sender = eidx;
    const int* recv   = eidx + N_EDGES;

    // --- workspace carve-out
    auto align256 = [](size_t x) { return (x + 255) & ~(size_t)255; };
    char* wsb = (char*)d_ws;
    size_t o = 0;
    unsigned short* sc16 = (unsigned short*)(wsb + o); o = align256(o + (size_t)N_NODES * HID * 2);
    float*          agg  = (float*)(wsb + o);          o = align256(o + (size_t)N_NODES * HID * 4);
    const size_t W1FR_L = (size_t)NCH1 * 8 * 32 * 16;  // shorts per layer
    const size_t W2FR_L = (size_t)NCH2 * 8 * 32 * 16;
    unsigned short* w1fr  = (unsigned short*)(wsb + o); o = align256(o + 2 * W1FR_L * 2);
    unsigned short* w2fr  = (unsigned short*)(wsb + o); o = align256(o + 2 * W2FR_L * 2);
    unsigned short* nu1fr = (unsigned short*)(wsb + o); o = align256(o + W2FR_L * 2);
    unsigned short* nu2fr = (unsigned short*)(wsb + o); o = align256(o + W2FR_L * 2);

    // 1) out = h
    copy_h_kernel<<<(N_NODES * OUTW + 255) / 256, 256, 0, stream>>>(h, out, N_NODES * OUTW);

    // 2) fragment-ready bf16 weights
    frag_build_kernel<<<(NCH1 * 256 + 255) / 256, 256, 0, stream>>>(mp_w1, CAT, NCH1, w1fr);
    frag_build_kernel<<<(NCH1 * 256 + 255) / 256, 256, 0, stream>>>(mp_w1 + (size_t)CAT * HID, CAT, NCH1, w1fr + W1FR_L);
    frag_build_kernel<<<(NCH2 * 256 + 255) / 256, 256, 0, stream>>>(mp_w2, HID, NCH2, w2fr);
    frag_build_kernel<<<(NCH2 * 256 + 255) / 256, 256, 0, stream>>>(mp_w2 + (size_t)HID * HID, HID, NCH2, w2fr + W2FR_L);
    frag_build_kernel<<<(NCH2 * 256 + 255) / 256, 256, 0, stream>>>(nu_w1, HID, NCH2, nu1fr);
    frag_build_kernel<<<(NCH2 * 256 + 255) / 256, 256, 0, stream>>>(nu_w2, HID, NCH2, nu2fr);

    // 3) two message-passing rounds
    for (int l = 0; l < 2; ++l) {
        init_layer_kernel<<<(N_NODES * HID + 255) / 256, 256, 0, stream>>>(out, sc16, agg);
        edge_mp_kernel<<<N_EDGES / (16 * 8), 256, 0, stream>>>(
            sc16, sender, recv, elen,
            w1fr + (size_t)l * W1FR_L, w2fr + (size_t)l * W2FR_L,
            mp_b1 + (size_t)l * HID, mp_b2 + (size_t)l * HID, agg);
        writeback_kernel<<<(N_NODES * HID + 255) / 256, 256, 0, stream>>>(agg, out);
    }

    // 4) node update (residual MLP on scalars)
    init_layer_kernel<<<(N_NODES * HID + 255) / 256, 256, 0, stream>>>(out, sc16, agg);
    node_update_kernel<<<N_NODES / 16, 32, 0, stream>>>(sc16, nu1fr, nu2fr, nu_b1, nu_b2, out);
}
